// PerceiverAttention_8100308320711
// MI455X (gfx1250) — matrix operand order repaired and checked by reading, not compiled, not run on hardware
//
#include <hip/hip_runtime.h>
#include <hip/hip_bf16.h>

#define DIMV 1024
#define HEADS 16
#define DHEAD 64
#define INNER 1024
#define NBATCH 8
#define NSEQ 4096
#define MLAT 64
#define NKV (NSEQ + MLAT) /* 4160 */
#define KVW 2048
#define EPSV 1e-5f

typedef __bf16 bf16;
typedef __attribute__((ext_vector_type(16))) __bf16 v16bf;
typedef __attribute__((ext_vector_type(8)))  __bf16 v8bf;
typedef __attribute__((ext_vector_type(8)))  float  v8f;

static __device__ inline v16bf make_frag(v8bf lo, v8bf hi) {
  v16bf r;
#pragma unroll
  for (int e = 0; e < 8; ++e) { r[e] = lo[e]; r[8 + e] = hi[e]; }
  return r;
}

// ---- gfx1250 async global->LDS copy (ASYNCcnt-tracked, no VGPR round trip)
static __device__ inline void async_copy_b128(void* lds_ptr, const void* gptr) {
  unsigned lds_addr = (unsigned)(unsigned long long)lds_ptr;  // low 32 bits = LDS offset
  unsigned long long gaddr = (unsigned long long)gptr;
  asm volatile("global_load_async_to_lds_b128 %0, %1, off"
               :: "v"(lds_addr), "v"(gaddr)
               : "memory");
}
template <int N>
static __device__ inline void wait_asynccnt() {
  asm volatile("s_wait_asynccnt %0" :: "i"(N) : "memory");
}

// ---------------------------------------------------------------- transpose f32[K][N] -> bf16[N][K]
__global__ __launch_bounds__(256) void transpose_to_bf16_kernel(
    const float* __restrict__ in, bf16* __restrict__ out, int K, int N) {
  __shared__ float tile[32][33];
  int tx = threadIdx.x & 31, ty = threadIdx.x >> 5;  // 32 x 8
  int k0 = blockIdx.y * 32, n0 = blockIdx.x * 32;
#pragma unroll
  for (int r = 0; r < 32; r += 8)
    tile[ty + r][tx] = in[(size_t)(k0 + ty + r) * N + n0 + tx];
  __syncthreads();
#pragma unroll
  for (int r = 0; r < 32; r += 8)
    out[(size_t)(n0 + ty + r) * K + k0 + tx] = (bf16)tile[tx][ty + r];
}

// ---------------------------------------------------------------- LayerNorm (row = 1024)
template <typename OutT>
__global__ __launch_bounds__(256) void layernorm_kernel(
    const float* __restrict__ in, OutT* __restrict__ out,
    const float* __restrict__ g, const float* __restrict__ bta,
    int rows_per_batch, int out_batch_stride, int out_row_offset) {
  __shared__ float s_sum[256];
  __shared__ float s_sq[256];
  int row = blockIdx.x;
  int b = row / rows_per_batch;
  int r = row - b * rows_per_batch;
  int t = threadIdx.x;
  const float* p = in + (size_t)row * DIMV;
  float4 v = *(const float4*)(p + t * 4);
  s_sum[t] = v.x + v.y + v.z + v.w;
  s_sq[t]  = v.x * v.x + v.y * v.y + v.z * v.z + v.w * v.w;
  __syncthreads();
#pragma unroll
  for (int off = 128; off > 0; off >>= 1) {
    if (t < off) { s_sum[t] += s_sum[t + off]; s_sq[t] += s_sq[t + off]; }
    __syncthreads();
  }
  float mu   = s_sum[0] * (1.0f / DIMV);
  float var  = s_sq[0] * (1.0f / DIMV) - mu * mu;
  float rstd = rsqrtf(var + EPSV);
  float4 gg = *(const float4*)(g + t * 4);
  float4 bb = *(const float4*)(bta + t * 4);
  OutT* o = out + ((size_t)b * out_batch_stride + out_row_offset + r) * DIMV + t * 4;
  o[0] = (OutT)((v.x - mu) * rstd * gg.x + bb.x);
  o[1] = (OutT)((v.y - mu) * rstd * gg.y + bb.y);
  o[2] = (OutT)((v.z - mu) * rstd * gg.z + bb.z);
  o[3] = (OutT)((v.w - mu) * rstd * gg.w + bb.w);
}

// ---------------------------------------------------------------- bf16 WMMA GEMM
// C[M,N] = alpha * A[M,K] @ B[K,N], with B supplied pre-transposed Bt[N][K].
// 128x64 tile per block, 128 threads (4 waves); wave owns two 16-row stripes.
// Double-buffered LDS fed by global_load_async_to_lds_b128 (ASYNCcnt).
template <typename OutT>
__global__ __launch_bounds__(128) void gemm_bf16_kernel(
    const bf16* __restrict__ A, const bf16* __restrict__ Bt,
    OutT* __restrict__ C, int M, int N, int K, float alpha) {
  __shared__ __align__(16) bf16 As[2][128][32];  // [m][k]
  __shared__ __align__(16) bf16 Bs[2][64][32];   // [n][k]
  int n0 = blockIdx.x * 64, m0 = blockIdx.y * 128;
  int tid = threadIdx.x, lane = tid & 31, w = tid >> 5;
  int c = lane & 15, lh = lane >> 4;

  auto stage = [&](int buf, int k0) {
#pragma unroll
    for (int it = 0; it < 4; ++it) {             // A tile: 128x32
      int linear = (tid + it * 128) * 8;
      int am = linear >> 5, ak = linear & 31;
      async_copy_b128(&As[buf][am][ak], A + (size_t)(m0 + am) * K + k0 + ak);
    }
#pragma unroll
    for (int it = 0; it < 2; ++it) {             // B tile: 64x32 (K-contiguous rows of Bt)
      int linear = (tid + it * 128) * 8;
      int bn = linear >> 5, bk = linear & 31;
      async_copy_b128(&Bs[buf][bn][bk], Bt + (size_t)(n0 + bn) * K + k0 + bk);
    }
  };

  v8f acc[2][4] = {};
  int nsteps = K / 32;
  stage(0, 0);
  for (int s = 0; s < nsteps; ++s) {
    int cur = s & 1;
    if (s + 1 < nsteps) {
      stage(cur ^ 1, (s + 1) * 32);
      wait_asynccnt<6>();                        // stage s complete; s+1 in flight
    } else {
      wait_asynccnt<0>();
    }
    __syncthreads();
    v16bf bfrag[4];
#pragma unroll
    for (int t = 0; t < 4; ++t) {
      int bc = t * 16 + c;
      bfrag[t] = make_frag(*(const v8bf*)&Bs[cur][bc][lh * 8],
                           *(const v8bf*)&Bs[cur][bc][16 + lh * 8]);
    }
#pragma unroll
    for (int half = 0; half < 2; ++half) {
      int ar = half * 64 + w * 16 + c;
      v16bf afrag = make_frag(*(const v8bf*)&As[cur][ar][lh * 8],
                              *(const v8bf*)&As[cur][ar][16 + lh * 8]);
#pragma unroll
      for (int t = 0; t < 4; ++t)
        acc[half][t] = __builtin_amdgcn_wmma_f32_16x16x32_bf16(
            false, afrag, false, bfrag[t], (short)0, acc[half][t], false, false);
    }
    __syncthreads();                             // everyone done with buffer `cur`
  }
#pragma unroll
  for (int half = 0; half < 2; ++half)
#pragma unroll
    for (int t = 0; t < 4; ++t) {
      int col = n0 + t * 16 + c;
#pragma unroll
      for (int i = 0; i < 8; ++i) {
        int row = m0 + half * 64 + w * 16 + lh * 8 + i;
        C[(size_t)row * N + col] = (OutT)(acc[half][t][i] * alpha);
      }
    }
}

// ---------------------------------------------------------------- flash attention
// One block per (batch, head).  128 threads = 4 waves; wave w owns query rows
// [w*16, w*16+16).  Iterates KV in 64-wide chunks with online softmax.
__global__ __launch_bounds__(128) void attn_kernel(
    const bf16* __restrict__ q,       // [8*64, 1024]  (already * 1/8)
    const bf16* __restrict__ kv,      // [8*4160, 2048]  (k | v)
    const unsigned char* __restrict__ mask,  // [8, 4096]
    bf16* __restrict__ outp) {        // [8*64, 1024]  layout [b, m, h, d]
  __shared__ __align__(16) bf16 Ks[64][64];    // [j][d]
  __shared__ __align__(16) bf16 Vt[64][64];    // [d][j]
  __shared__ __align__(16) bf16 Ps[4][16][64]; // per-wave P tile [row][j]
  int bh = blockIdx.x;
  int b = bh >> 4, hd = bh & 15;
  int tid = threadIdx.x, lane = tid & 31, w = tid >> 5;
  int c = lane & 15, lh = lane >> 4;

  const bf16* qrow = q + (size_t)(b * MLAT + w * 16 + c) * INNER + hd * DHEAD;
  v16bf aq[2];
#pragma unroll
  for (int f = 0; f < 2; ++f)
    aq[f] = make_frag(*(const v8bf*)(qrow + f * 32 + lh * 8),
                      *(const v8bf*)(qrow + f * 32 + 16 + lh * 8));

  float rmax[8], rsum[8];
  v8f oacc[4] = {};
#pragma unroll
  for (int i = 0; i < 8; ++i) { rmax[i] = -3.0e38f; rsum[i] = 0.f; }

  const bf16* kbase = kv + (size_t)b * NKV * KVW + hd * DHEAD;
  const bf16* vbase = kbase + INNER;
  const unsigned char* mk = mask + (size_t)b * NSEQ;

  for (int j0 = 0; j0 < NKV; j0 += 64) {
    __syncthreads();
#pragma unroll
    for (int it = 0; it < 4; ++it) {   // stage K (async) and V (transposed, via VGPRs)
      int linear = tid + it * 128;
      int j = linear >> 3, dv = (linear & 7) * 8;
      size_t roff = (size_t)(j0 + j) * KVW + dv;
      async_copy_b128(&Ks[j][dv], kbase + roff);
      v8bf vv = *(const v8bf*)(vbase + roff);
#pragma unroll
      for (int e = 0; e < 8; ++e) Vt[dv + e][j] = vv[e];
    }
    wait_asynccnt<0>();
    __syncthreads();

    // ---- S = q @ K^T  (16 rows x 64 cols per wave)
    v8f sacc[4] = {};
#pragma unroll
    for (int t = 0; t < 4; ++t) {
      int jc = t * 16 + c;
#pragma unroll
      for (int f = 0; f < 2; ++f) {
        v16bf bk = make_frag(*(const v8bf*)&Ks[jc][f * 32 + lh * 8],
                             *(const v8bf*)&Ks[jc][f * 32 + 16 + lh * 8]);
        sacc[t] = __builtin_amdgcn_wmma_f32_16x16x32_bf16(
            false, aq[f], false, bk, (short)0, sacc[t], false, false);
      }
    }

    // ---- mask + online softmax (row reductions over 16 lanes of a half-wave)
    bool mok[4];
#pragma unroll
    for (int t = 0; t < 4; ++t) {
      int jg = j0 + t * 16 + c;
      mok[t] = (jg >= NSEQ) ? true : (mk[jg] != 0);
    }
    float p[4][8];
#pragma unroll
    for (int i = 0; i < 8; ++i) {
      float mx = -3.0e38f;
#pragma unroll
      for (int t = 0; t < 4; ++t) {
        float s = mok[t] ? sacc[t][i] : -3.0e38f;
        p[t][i] = s;
        mx = fmaxf(mx, s);
      }
#pragma unroll
      for (int off = 1; off < 16; off <<= 1)
        mx = fmaxf(mx, __shfl_xor(mx, off, 16));
      float nm = fmaxf(rmax[i], mx);
      float sc = __expf(rmax[i] - nm);
      rmax[i] = nm;
      float rs = 0.f;
#pragma unroll
      for (int t = 0; t < 4; ++t) {
        float pv = __expf(p[t][i] - nm);
        p[t][i] = pv;
        rs += pv;
      }
#pragma unroll
      for (int off = 1; off < 16; off <<= 1)
        rs += __shfl_xor(rs, off, 16);
      rsum[i] = rsum[i] * sc + rs;
#pragma unroll
      for (int t = 0; t < 4; ++t) oacc[t][i] *= sc;
    }

    // ---- re-layout P (D-fragment -> A-fragment) via per-wave LDS tile
#pragma unroll
    for (int t = 0; t < 4; ++t)
#pragma unroll
      for (int i = 0; i < 8; ++i)
        Ps[w][lh * 8 + i][t * 16 + c] = (bf16)p[t][i];

    // ---- O += P @ V
#pragma unroll
    for (int f = 0; f < 2; ++f) {
      v16bf aP = make_frag(*(const v8bf*)&Ps[w][c][f * 32 + lh * 8],
                           *(const v8bf*)&Ps[w][c][f * 32 + 16 + lh * 8]);
#pragma unroll
      for (int t = 0; t < 4; ++t) {
        int dc = t * 16 + c;
        v16bf bv = make_frag(*(const v8bf*)&Vt[dc][f * 32 + lh * 8],
                             *(const v8bf*)&Vt[dc][f * 32 + 16 + lh * 8]);
        oacc[t] = __builtin_amdgcn_wmma_f32_16x16x32_bf16(
            false, aP, false, bv, (short)0, oacc[t], false, false);
      }
    }
  }

  bf16* orow = outp + (size_t)(b * MLAT) * INNER + hd * DHEAD;
#pragma unroll
  for (int i = 0; i < 8; ++i) {
    float inv = 1.0f / rsum[i];
    int row = w * 16 + lh * 8 + i;
#pragma unroll
    for (int t = 0; t < 4; ++t)
      orow[(size_t)row * INNER + t * 16 + c] = (bf16)(oacc[t][i] * inv);
  }
}

// ---------------------------------------------------------------- launch
extern "C" void kernel_launch(void* const* d_in, const int* in_sizes, int n_in,
                              void* d_out, int out_size, void* d_ws, size_t ws_size,
                              hipStream_t stream) {
  (void)in_sizes; (void)n_in; (void)out_size; (void)ws_size;
  const float* x      = (const float*)d_in[0];
  const float* lat    = (const float*)d_in[1];
  const unsigned char* mask = (const unsigned char*)d_in[2];
  const float* ln_x_g = (const float*)d_in[3];
  const float* ln_x_b = (const float*)d_in[4];
  const float* ln_l_g = (const float*)d_in[5];
  const float* ln_l_b = (const float*)d_in[6];
  const float* Wq     = (const float*)d_in[7];
  const float* Wkv    = (const float*)d_in[8];
  const float* Wout   = (const float*)d_in[9];
  const float* ln_o_g = (const float*)d_in[10];
  const float* ln_o_b = (const float*)d_in[11];

  char* ws = (char*)d_ws;
  size_t off = 0;
  auto alloc = [&](size_t bytes) -> void* {
    void* p = ws + off;
    off = (off + bytes + 255) & ~(size_t)255;
    return p;
  };
  bf16* WqT   = (bf16*)alloc((size_t)DIMV * INNER * 2);      // [INNER][DIMV]
  bf16* WkvT  = (bf16*)alloc((size_t)KVW * DIMV * 2);        // [KVW][DIMV]
  bf16* WoutT = (bf16*)alloc((size_t)DIMV * INNER * 2);      // [DIMV][INNER]
  bf16* lnq    = (bf16*)alloc((size_t)NBATCH * MLAT * DIMV * 2);
  bf16* qb     = (bf16*)alloc((size_t)NBATCH * MLAT * INNER * 2);
  bf16* xnln   = (bf16*)alloc((size_t)NBATCH * NKV * DIMV * 2);
  bf16* kvb    = (bf16*)alloc((size_t)NBATCH * NKV * KVW * 2);
  bf16* attn_o = (bf16*)alloc((size_t)NBATCH * MLAT * INNER * 2);
  float* out_f = (float*)alloc((size_t)NBATCH * MLAT * DIMV * 4);

  // 1) weights -> bf16, transposed to [N][K] so GEMM B-tiles are K-contiguous
  {
    dim3 g1(INNER / 32, DIMV / 32);
    transpose_to_bf16_kernel<<<g1, 256, 0, stream>>>(Wq, WqT, DIMV, INNER);
    dim3 g2(KVW / 32, DIMV / 32);
    transpose_to_bf16_kernel<<<g2, 256, 0, stream>>>(Wkv, WkvT, DIMV, KVW);
    dim3 g3(DIMV / 32, INNER / 32);
    transpose_to_bf16_kernel<<<g3, 256, 0, stream>>>(Wout, WoutT, INNER, DIMV);
  }
  // 2) LayerNorms -> bf16
  layernorm_kernel<bf16><<<NBATCH * NSEQ, 256, 0, stream>>>(
      x, xnln, ln_x_g, ln_x_b, NSEQ, NKV, 0);
  layernorm_kernel<bf16><<<NBATCH * MLAT, 256, 0, stream>>>(
      lat, xnln, ln_l_g, ln_l_b, MLAT, NKV, NSEQ);
  layernorm_kernel<bf16><<<NBATCH * MLAT, 256, 0, stream>>>(
      lat, lnq, ln_l_g, ln_l_b, MLAT, MLAT, 0);
  // 3) q = lnq @ Wq  (alpha = dim_head^-0.5 = 1/8)
  {
    dim3 grid(INNER / 64, (NBATCH * MLAT) / 128);
    gemm_bf16_kernel<bf16><<<grid, 128, 0, stream>>>(
        lnq, WqT, qb, NBATCH * MLAT, INNER, DIMV, 0.125f);
  }
  // 4) kv = concat(xn, ln) @ Wkv
  {
    dim3 grid(KVW / 64, (NBATCH * NKV) / 128);
    gemm_bf16_kernel<bf16><<<grid, 128, 0, stream>>>(
        xnln, WkvT, kvb, NBATCH * NKV, KVW, DIMV, 1.0f);
  }
  // 5) flash attention
  attn_kernel<<<NBATCH * HEADS, 128, 0, stream>>>(qb, kvb, mask, attn_o);
  // 6) out = attn_o @ Wout  (f32 result)
  {
    dim3 grid(DIMV / 64, (NBATCH * MLAT) / 128);
    gemm_bf16_kernel<float><<<grid, 128, 0, stream>>>(
        attn_o, WoutT, out_f, NBATCH * MLAT, DIMV, INNER, 1.0f);
  }
  // 7) final LayerNorm -> d_out (f32)
  layernorm_kernel<float><<<NBATCH * MLAT, 256, 0, stream>>>(
      out_f, (float*)d_out, ln_o_g, ln_o_b, NBATCH * MLAT, NBATCH * MLAT, 0);
}
